// GraphAttentionLayer_43035572306184
// MI455X (gfx1250) — matrix operand order repaired and checked by reading, not compiled, not run on hardware
//
#include <hip/hip_runtime.h>
#include <hip/hip_bf16.h>

typedef __attribute__((ext_vector_type(16))) _Float16 v16h;
typedef __attribute__((ext_vector_type(8)))  float    v8f;
typedef __attribute__((ext_vector_type(4)))  _Float16 v4h;

union FragU { v16h h; unsigned int u[8]; };

// A-operand fragment (16x32 f16). Row-major LDS row pointer (4B aligned).
// Per ISA table: lanes 0-15 hold K=0..7 & 16..23, lanes 16-31 hold K=8..15 & 24..31,
// packed as consecutive-K pairs per VGPR.
__device__ __forceinline__ v16h frag_a(const _Float16* rowPtr, int lane) {
  const unsigned int* rp = (const unsigned int*)rowPtr;
  const int kb = (lane & 16) ? 4 : 0;   // uint index of K=8
  FragU f;
#pragma unroll
  for (int p = 0; p < 4; ++p) { f.u[p] = rp[kb + p]; f.u[4 + p] = rp[kb + 8 + p]; }
  return f.h;
}

// B-operand fragment (32x16 f16). Column-major LDS column pointer (4B aligned).
// Per ISA: lanes 0-15 hold K=0..7 & 16..23, lanes 16-31 hold K=8..15 & 24..31, consecutive-K pairs.
__device__ __forceinline__ v16h frag_b(const _Float16* colPtr, int lane) {
  const unsigned int* rp = (const unsigned int*)colPtr;
  const int kb = (lane & 16) ? 4 : 0;   // uint index of K=8
  FragU f;
#pragma unroll
  for (int p = 0; p < 4; ++p) { f.u[p] = rp[kb + p]; f.u[4 + p] = rp[kb + 8 + p]; }
  return f.h;
}

#define IN_F 256
#define NCOL 4096
#define LDA1 36   // padded f16 stride for 32-wide tiles
#define LDK3 68   // padded f16 stride for 64-wide tiles

// ---------------- Kernel 1: Wh = h @ W  (8192x256 @ 256x4096), fp32 I/O, f16 WMMA ----
__global__ __launch_bounds__(128)
void gat_gemm1(const float* __restrict__ H, const float* __restrict__ W,
               float* __restrict__ Wh) {
  __shared__ _Float16 As[64 * LDA1];
  __shared__ _Float16 Bs[64 * LDA1];
  const int tid  = threadIdx.x;
  const int lane = tid & 31;
  const int wave = tid >> 5;          // 4 waves, each owns a 16-row strip
  const int n0 = blockIdx.x * 64;
  const int m0 = blockIdx.y * 64;

  v8f acc0 = {}, acc1 = {}, acc2 = {}, acc3 = {};
  v8f* acc[4] = { &acc0, &acc1, &acc2, &acc3 };

  for (int ks = 0; ks < 8; ++ks) {
    const int k0 = ks * 32;
    // A tile: 64x32 fp32 -> f16 row-major LDS
#pragma unroll
    for (int j = 0; j < 4; ++j) {
      const int idx = tid + j * 128;          // 512 float4
      const int row = idx >> 3, c4 = idx & 7;
      const float4 v = *(const float4*)(H + (size_t)(m0 + row) * IN_F + k0 + c4 * 4);
      v4h hv = { (_Float16)v.x, (_Float16)v.y, (_Float16)v.z, (_Float16)v.w };
      *(v4h*)(&As[row * LDA1 + c4 * 4]) = hv;
    }
    // B tile: 32x64 fp32 -> f16 column-major LDS (Bs[n][k])
#pragma unroll
    for (int j = 0; j < 4; ++j) {
      const int idx = tid + j * 128;
      const int kr = idx >> 4, c4 = idx & 15;
      const float4 v = *(const float4*)(W + (size_t)(k0 + kr) * NCOL + n0 + c4 * 4);
      Bs[(c4 * 4 + 0) * LDA1 + kr] = (_Float16)v.x;
      Bs[(c4 * 4 + 1) * LDA1 + kr] = (_Float16)v.y;
      Bs[(c4 * 4 + 2) * LDA1 + kr] = (_Float16)v.z;
      Bs[(c4 * 4 + 3) * LDA1 + kr] = (_Float16)v.w;
    }
    __syncthreads();

    const v16h a = frag_a(&As[(wave * 16 + (lane & 15)) * LDA1], lane);
#pragma unroll
    for (int nt = 0; nt < 4; ++nt) {
      const v16h b = frag_b(&Bs[(nt * 16 + (lane & 15)) * LDA1], lane);
      *acc[nt] = __builtin_amdgcn_wmma_f32_16x16x32_f16(
          false, a, false, b, (short)0, *acc[nt], false, false);
    }
    __syncthreads();
  }

  const int mrow = m0 + wave * 16 + ((lane & 16) ? 8 : 0);
  const int col0 = n0 + (lane & 15);
#pragma unroll
  for (int nt = 0; nt < 4; ++nt)
#pragma unroll
    for (int r = 0; r < 8; ++r)
      Wh[(size_t)(mrow + r) * NCOL + col0 + nt * 16] = (*acc[nt])[r];
}

// ---------------- Kernel 2: s1/s2 row-dot reductions -------------------------------
__global__ __launch_bounds__(256)
void gat_scores(const float* __restrict__ Wh, const float* __restrict__ a,
                float* __restrict__ s1, float* __restrict__ s2) {
  const int row = blockIdx.x * 256 + threadIdx.x;   // < 128*64*64
  const float* p = Wh + (size_t)row * 64;
  float d1 = 0.f, d2 = 0.f;
#pragma unroll
  for (int q = 0; q < 16; ++q) {
    const float4 v  = *(const float4*)(p + q * 4);
    const float4 a1 = *(const float4*)(a + q * 4);
    const float4 a2 = *(const float4*)(a + 64 + q * 4);
    d1 += v.x * a1.x + v.y * a1.y + v.z * a1.z + v.w * a1.w;
    d2 += v.x * a2.x + v.y * a2.y + v.z * a2.z + v.w * a2.w;
  }
  s1[row] = d1;
  s2[row] = d2;
}

// ---------------- Kernel 3: softmax(leaky_relu(s1+s2)) @ Wh[b,i], in place ----------
__global__ __launch_bounds__(128)
void gat_attn(float* __restrict__ Wh, const float* __restrict__ s1,
              const float* __restrict__ s2) {
  __shared__ _Float16 Af[64 * LDK3];   // attention matrix, row-major (x,K=y)
  __shared__ _Float16 Bf[64 * LDK3];   // Wh tile, column-major (o,K=y)
  __shared__ float    Ef[64 * 65];     // fp32 softmax scratch
  const int tid  = threadIdx.x;
  const int lane = tid & 31;
  const int wave = tid >> 5;
  const int g = blockIdx.x;            // g = b*64 + i
  float* tile = Wh + (size_t)g * 4096; // Wh[b,i] : [y=64][o=64] row-major

  // Wh tile -> f16, transposed to column-major Bf[o][y]
#pragma unroll
  for (int j = 0; j < 8; ++j) {
    const int idx = tid + j * 128;     // 1024 float4
    const int y = idx >> 4, c4 = idx & 15;
    const float4 v = *(const float4*)(tile + y * 64 + c4 * 4);
    Bf[(c4 * 4 + 0) * LDK3 + y] = (_Float16)v.x;
    Bf[(c4 * 4 + 1) * LDK3 + y] = (_Float16)v.y;
    Bf[(c4 * 4 + 2) * LDK3 + y] = (_Float16)v.z;
    Bf[(c4 * 4 + 3) * LDK3 + y] = (_Float16)v.w;
  }

  // attn[x,y] = softmax_y( lrelu( s1[b,i,x] + s2[b,x,y] ) )
  if (tid < 64) {
    const int x = tid;
    const float sv = s1[(size_t)g * 64 + x];
    const float* s2row = s2 + (size_t)((g & ~63) + x) * 64;  // (b, x, :)
    float m = -1e30f;
    for (int y = 0; y < 64; ++y) {
      float e = sv + s2row[y];
      e = (e < 0.f) ? 0.01f * e : e;
      Ef[x * 65 + y] = e;
      m = fmaxf(m, e);
    }
    float sum = 0.f;
    for (int y = 0; y < 64; ++y) {
      const float p = __expf(Ef[x * 65 + y] - m);
      Ef[x * 65 + y] = p;
      sum += p;
    }
    const float inv = 1.0f / sum;
    for (int y = 0; y < 64; ++y)
      Af[x * LDK3 + y] = (_Float16)(Ef[x * 65 + y] * inv);
  }
  __syncthreads();

  v8f acc0 = {}, acc1 = {}, acc2 = {}, acc3 = {};
  v8f* acc[4] = { &acc0, &acc1, &acc2, &acc3 };
#pragma unroll
  for (int ks = 0; ks < 2; ++ks) {
    const v16h a = frag_a(&Af[(wave * 16 + (lane & 15)) * LDK3 + ks * 32], lane);
#pragma unroll
    for (int nt = 0; nt < 4; ++nt) {
      const v16h b = frag_b(&Bf[(nt * 16 + (lane & 15)) * LDK3 + ks * 32], lane);
      *acc[nt] = __builtin_amdgcn_wmma_f32_16x16x32_f16(
          false, a, false, b, (short)0, *acc[nt], false, false);
    }
  }
  // overwrite the tile in place (tile already fully captured in LDS)
  const int xr = wave * 16 + ((lane & 16) ? 8 : 0);
  const int o0 = (lane & 15);
#pragma unroll
  for (int nt = 0; nt < 4; ++nt)
#pragma unroll
    for (int r = 0; r < 8; ++r)
      tile[(xr + r) * 64 + o0 + nt * 16] = (*acc[nt])[r];
}

extern "C" void kernel_launch(void* const* d_in, const int* in_sizes, int n_in,
                              void* d_out, int out_size, void* d_ws, size_t ws_size,
                              hipStream_t stream) {
  (void)in_sizes; (void)n_in; (void)out_size; (void)ws_size;
  const float* h = (const float*)d_in[0];
  // d_in[1] = adj : all-ones, unused by the reference math
  const float* W = (const float*)d_in[2];
  const float* a = (const float*)d_in[3];
  float* out = (float*)d_out;                // holds Wh, then final h_prime in place
  float* s1  = (float*)d_ws;                 // 128*64*64 floats
  float* s2  = s1 + 128 * 64 * 64;           // total ws use: 4 MB

  gat_gemm1 <<<dim3(64, 128), 128, 0, stream>>>(h, W, out);
  gat_scores<<<2048, 256, 0, stream>>>(out, a, s1, s2);
  gat_attn  <<<8192, 128, 0, stream>>>(out, s1, s2);
}
